// RG_SNN_PureOriginal_42382737277006
// MI455X (gfx1250) — matrix operand order repaired and checked by reading, not compiled, not run on hardware
//
#include <hip/hip_runtime.h>
#include <hip/hip_bf16.h>

// ---------------------------------------------------------------------------
// CDNA5 / gfx1250 implementation of the SNN reference.
// Convolutions run as bf16 WMMA GEMMs (v_wmma_f32_16x16x32_bf16, f32 acc).
// conv2/conv3 stage their shared spike A-tiles into LDS with double-buffered
// GLOBAL_LOAD_ASYNC_TO_LDS_B128 (ASYNCcnt) when the builtin is available.
// Spikes are exact in bf16; weights are converted once per call.
// ---------------------------------------------------------------------------

typedef __attribute__((ext_vector_type(16))) __bf16 v16bf;
typedef __attribute__((ext_vector_type(8)))  __bf16 v8bf;
typedef __attribute__((ext_vector_type(8)))  float  v8f;
typedef __attribute__((ext_vector_type(4)))  int    v4i;

#define B_SZ   128
#define T_SZ   480
#define TPAD   864     // 192 + 480 + 192
#define PADLO  192

#if defined(__has_builtin)
#  if __has_builtin(__builtin_amdgcn_global_load_async_to_lds_b128) && \
      __has_builtin(__builtin_amdgcn_s_wait_asynccnt)
#    define USE_ASYNC 1
#  endif
#endif
#ifndef USE_ASYNC
#  define USE_ASYNC 0
#endif

__device__ __forceinline__ __bf16 f2bf(float f) {
  union { float f; unsigned u; } v; v.f = f;
  unsigned r = v.u + 0x7FFFu + ((v.u >> 16) & 1u);   // round-to-nearest-even
  unsigned short h = (unsigned short)(r >> 16);
  __bf16 out;
  __builtin_memcpy(&out, &h, 2);
  return out;
}

__device__ __forceinline__ v16bf pack16(v8bf lo, v8bf hi) {
  v16bf r;
#pragma unroll
  for (int e = 0; e < 8; ++e) { r[e] = lo[e]; r[8 + e] = hi[e]; }
  return r;
}

// 16-byte global -> LDS copy. Async (ASYNCcnt-tracked, no VGPR round trip)
// when the toolchain exposes the gfx1250 builtin; plain copy otherwise.
__device__ __forceinline__ void copy16_g2l(const __bf16* g, __bf16* l) {
#if USE_ASYNC
  // Builtin signature (from hipcc diagnostic): (v4i AS1*, v4i AS3*, Ii, Ii).
  // Global addresses are flat -> AS1 value == generic value.
  // Generic LDS pointers carry the LDS offset in their low 32 bits
  // so integer truncation yields the
  // 32-bit AS3 pointer.
  __attribute__((address_space(1))) v4i* gp =
      (__attribute__((address_space(1))) v4i*)(unsigned long long)(const void*)g;
  __attribute__((address_space(3))) v4i* lp =
      (__attribute__((address_space(3))) v4i*)(unsigned int)(unsigned long long)(void*)l;
  __builtin_amdgcn_global_load_async_to_lds_b128(gp, lp, 0, 0);
#else
  *(uint4*)l = *(const uint4*)g;
#endif
}

__device__ __forceinline__ void async_wait_all() {
#if USE_ASYNC
  __builtin_amdgcn_s_wait_asynccnt(0);
#endif
}

// --------------------------- utility kernels -------------------------------

__global__ void zero_kernel(uint4* __restrict__ p, int n16) {
  for (int i = blockIdx.x * blockDim.x + threadIdx.x; i < n16;
       i += gridDim.x * blockDim.x) {
    p[i] = make_uint4(0u, 0u, 0u, 0u);
  }
}

// Convert + transpose weights to bf16 "B-matrix friendly" layout [tap][co][ci]
__global__ void prep_weights(const float* __restrict__ w1,   // (w=64, co=64)
                             const float* __restrict__ w2,   // (k=32, ci=64, co=128)
                             const float* __restrict__ w3,   // (k=32, ci=128, co=128)
                             __bf16* __restrict__ w1t,       // [co][w]
                             __bf16* __restrict__ w2t,       // [k][co][ci]
                             __bf16* __restrict__ w3t) {     // [k][co][ci]
  int i = blockIdx.x * blockDim.x + threadIdx.x;
  if (i < 64 * 64) {
    int w = i >> 6, co = i & 63;
    w1t[co * 64 + w] = f2bf(w1[i]);
  }
  if (i < 32 * 64 * 128) {
    int k = i / (64 * 128); int r = i % (64 * 128);
    int ci = r >> 7, co = r & 127;
    w2t[((size_t)k * 128 + co) * 64 + ci] = f2bf(w2[i]);
  }
  if (i < 32 * 128 * 128) {
    int k = i / (128 * 128); int r = i % (128 * 128);
    int ci = r >> 7, co = r & 127;
    w3t[((size_t)k * 128 + co) * 128 + ci] = f2bf(w3[i]);
  }
}

// --------------------------- conv1 (GEMM 61440x64x64) ----------------------
// One wave computes a 16(M) x 64(N) tile; 2 K-blocks of 32.
__global__ void conv1_bn_wmma(const float* __restrict__ x,
                              const __bf16* __restrict__ w1t,
                              const float* __restrict__ sc_, const float* __restrict__ bi_,
                              const float* __restrict__ mu_, const float* __restrict__ var_,
                              float* __restrict__ x1) {
  const int wave = (blockIdx.x * blockDim.x + threadIdx.x) >> 5;  // 0..3839
  const int lane = threadIdx.x & 31;
  const int half = lane >> 4;
  const int m    = lane & 15;
  const int row0 = wave * 16;                                     // row in B*T

  v8f acc[4] = {v8f{}, v8f{}, v8f{}, v8f{}};

#pragma unroll
  for (int kb = 0; kb < 64; kb += 32) {
    // A fragment: rows of x (f32 -> bf16 on the fly)
    const float* xr = x + (size_t)(row0 + m) * 64 + kb + half * 8;
    v16bf a;
#pragma unroll
    for (int j = 0; j < 8; ++j) a[j]     = f2bf(xr[j]);
#pragma unroll
    for (int j = 0; j < 8; ++j) a[8 + j] = f2bf(xr[16 + j]);

#pragma unroll
    for (int n = 0; n < 4; ++n) {
      const int co = n * 16 + m;
      const __bf16* wp = w1t + (size_t)co * 64 + kb + half * 8;
      v16bf bf = pack16(*(const v8bf*)wp, *(const v8bf*)(wp + 16));
      acc[n] = __builtin_amdgcn_wmma_f32_16x16x32_bf16(
          false, a, false, bf, (short)0, acc[n], false, false);
    }
  }

#pragma unroll
  for (int n = 0; n < 4; ++n) {
    const int c = n * 16 + m;
    const float sc = sc_[c] * rsqrtf(var_[c] + 1e-5f);
    const float bb = bi_[c] - mu_[c] * sc;
#pragma unroll
    for (int v = 0; v < 8; ++v) {
      const int M = half * 8 + v;
      x1[(size_t)(row0 + M) * 64 + c] = acc[n][v] * sc + bb;
    }
  }
}

// ------------------- conv2: 32-tap dilated(4) conv as WMMA -----------------
// Block = (b, t0): 8 waves cover co 0..127 (16 each); the per-tap 16x64 bf16
// A-tile is staged in LDS (double-buffered async copy shared by all waves).
__global__ void __launch_bounds__(256)
conv2_bn_wmma(const __bf16* __restrict__ spk1p,   // (B,864,64) bf16
              const __bf16* __restrict__ w2t,     // [k][co][ci=64]
              const float* __restrict__ sc_, const float* __restrict__ bi_,
              const float* __restrict__ mu_, const float* __restrict__ var_,
              float* __restrict__ x2) {            // (B,480,128)
  __shared__ __bf16 abuf[2][16 * 64];

  const int tid  = threadIdx.x;
  const int wave = tid >> 5;            // 0..7 -> co tile
  const int lane = tid & 31;
  const int half = lane >> 4;
  const int m    = lane & 15;
  const int t0   = (blockIdx.x % 30) * 16;
  const int b    = blockIdx.x / 30;
  const int co   = wave * 16 + m;

  // loader mapping: 128 threads x 16B cover the 2KB tile
  const int lrow   = (tid & 127) >> 3;        // 0..15
  const int lchunk = (tid & 7) * 8;           // element offset 0..56
  const __bf16* gA = spk1p +
      ((size_t)b * TPAD + (PADLO - 62 + t0 + lrow)) * 64 + lchunk;

  v8f acc = v8f{};

  // prologue: stage tap 0 into buffer 0
  if (tid < 128) copy16_g2l(gA, &abuf[0][lrow * 64 + lchunk]);
  async_wait_all();
  __syncthreads();

  for (int k = 0; k < 32; ++k) {
    const int cur = k & 1;
    if (k + 1 < 32 && tid < 128) {
      copy16_g2l(gA + (size_t)(4 * (k + 1)) * 64,
                 &abuf[1 - cur][lrow * 64 + lchunk]);
    }
#pragma unroll
    for (int cb = 0; cb < 2; ++cb) {
      const __bf16* ap = &abuf[cur][m * 64 + cb * 32 + half * 8];
      v16bf a = pack16(*(const v8bf*)ap, *(const v8bf*)(ap + 16));
      const __bf16* wp = w2t + ((size_t)k * 128 + co) * 64 + cb * 32 + half * 8;
      v16bf bf = pack16(*(const v8bf*)wp, *(const v8bf*)(wp + 16));
      acc = __builtin_amdgcn_wmma_f32_16x16x32_bf16(
          false, a, false, bf, (short)0, acc, false, false);
    }
    async_wait_all();
    __syncthreads();
  }

  const float sc = sc_[co] * rsqrtf(var_[co] + 1e-5f);
  const float bb = bi_[co] - mu_[co] * sc;
#pragma unroll
  for (int v = 0; v < 8; ++v) {
    const int M = half * 8 + v;
    x2[((size_t)b * T_SZ + t0 + M) * 128 + co] = acc[v] * sc + bb;
  }
}

// ------------------- conv3: 32-tap dilated(12) conv, K=128 -----------------
__global__ void __launch_bounds__(256)
conv3_bn_wmma(const __bf16* __restrict__ spk2p,   // (B,864,128) bf16
              const __bf16* __restrict__ w3t,     // [k][co][ci=128]
              const float* __restrict__ sc_, const float* __restrict__ bi_,
              const float* __restrict__ mu_, const float* __restrict__ var_,
              float* __restrict__ x3) {            // (B,480,128)
  __shared__ __bf16 abuf[2][16 * 128];

  const int tid  = threadIdx.x;
  const int wave = tid >> 5;            // 0..7 -> co tile
  const int lane = tid & 31;
  const int half = lane >> 4;
  const int m    = lane & 15;
  const int t0   = (blockIdx.x % 30) * 16;
  const int b    = blockIdx.x / 30;
  const int co   = wave * 16 + m;

  // loader mapping: 256 threads x 16B cover the 4KB tile
  const int lrow   = tid >> 4;                // 0..15
  const int lchunk = (tid & 15) * 8;          // 0..120
  const __bf16* gA = spk2p +
      ((size_t)b * TPAD + (PADLO - 186 + t0 + lrow)) * 128 + lchunk;

  v8f acc = v8f{};

  copy16_g2l(gA, &abuf[0][lrow * 128 + lchunk]);
  async_wait_all();
  __syncthreads();

  for (int k = 0; k < 32; ++k) {
    const int cur = k & 1;
    if (k + 1 < 32) {
      copy16_g2l(gA + (size_t)(12 * (k + 1)) * 128,
                 &abuf[1 - cur][lrow * 128 + lchunk]);
    }
#pragma unroll
    for (int cb = 0; cb < 4; ++cb) {
      const __bf16* ap = &abuf[cur][m * 128 + cb * 32 + half * 8];
      v16bf a = pack16(*(const v8bf*)ap, *(const v8bf*)(ap + 16));
      const __bf16* wp = w3t + ((size_t)k * 128 + co) * 128 + cb * 32 + half * 8;
      v16bf bf = pack16(*(const v8bf*)wp, *(const v8bf*)(wp + 16));
      acc = __builtin_amdgcn_wmma_f32_16x16x32_bf16(
          false, a, false, bf, (short)0, acc, false, false);
    }
    async_wait_all();
    __syncthreads();
  }

  const float sc = sc_[co] * rsqrtf(var_[co] + 1e-5f);
  const float bb = bi_[co] - mu_[co] * sc;
#pragma unroll
  for (int v = 0; v < 8; ++v) {
    const int M = half * 8 + v;
    x3[((size_t)b * T_SZ + t0 + M) * 128 + co] = acc[v] * sc + bb;
  }
}

// --------------------------- hLIF scan (stages 1 & 2) ----------------------
__global__ void hlif_kernel(const float* __restrict__ xin,
                            const float* __restrict__ vth_raw,
                            const float* __restrict__ decay_raw,
                            float* __restrict__ spk_out,    // f32, part of d_out
                            __bf16* __restrict__ spk_pad,   // bf16 padded, for next conv
                            int C) {
  const int tid = blockIdx.x * blockDim.x + threadIdx.x;
  const int c = tid % C;
  const int b = tid / C;
  if (b >= B_SZ) return;

  // v_m=0, v_s=1, d_m=2, d_s=1 for both stages
  const float vth = log1pf(expf(vth_raw[c])) + 0.5f;
  float dec = 1.0f / (1.0f + expf(-(decay_raw[c] + 2.0f)));
  dec = fminf(fmaxf(dec, 0.0f), 0.99f);

  const float* xp = xin     + (size_t)b * T_SZ * C + c;
  float*       so = spk_out + (size_t)b * T_SZ * C + c;
  __bf16*      sp = spk_pad + ((size_t)b * TPAD + PADLO) * C + c;

  float v = 0.0f;
  for (int t = 0; t < T_SZ; ++t) {
    v = v * dec + xp[(size_t)t * C];
    const float s = (v - vth) > 0.0f ? 1.0f : 0.0f;
    v -= s * vth;
    so[(size_t)t * C] = s;
    sp[(size_t)t * C] = f2bf(s);
  }
}

// --------------------------- ALIF scan (stage 3) ---------------------------
__global__ void alif_kernel(const float* __restrict__ xin,
                            float* __restrict__ spk_out) {   // (B,480,128) f32
  const int tid = blockIdx.x * blockDim.x + threadIdx.x;
  const int c = tid & 127;
  const int b = tid >> 7;
  if (b >= B_SZ) return;

  const float d = 0.9f, adp = 0.9f, beta = 1.8f;
  const float* xp = xin     + (size_t)b * T_SZ * 128 + c;
  float*       so = spk_out + (size_t)b * T_SZ * 128 + c;

  float v = 0.0f, th = 0.0f, ps = 0.0f;
  for (int t = 0; t < T_SZ; ++t) {
    th = th * adp + ps * beta;
    v  = v * d + xp[(size_t)t * 128];
    const float vth = 0.5f + th;
    const float s = (v - vth) > 0.0f ? 1.0f : 0.0f;
    v -= s * vth;
    ps = s;
    so[(size_t)t * 128] = s;
  }
}

// --------------------------- fused readout ---------------------------------
__global__ void readout_kernel(const float* __restrict__ spk3,
                               const float* __restrict__ dw, const float* __restrict__ db,
                               const float* __restrict__ a1w, const float* __restrict__ a1b,
                               const float* __restrict__ a2w, const float* __restrict__ a2b,
                               float* __restrict__ out) {
  __shared__ float den[T_SZ * 4];
  __shared__ float dp[20 * 4];
  __shared__ float score[20];

  const int b = blockIdx.x;
  const int tid = threadIdx.x;

  for (int t = tid; t < T_SZ; t += blockDim.x) {
    const float* xr = spk3 + ((size_t)b * T_SZ + t) * 128;
    float o0 = db[0], o1 = db[1], o2 = db[2], o3 = db[3];
    for (int c = 0; c < 128; ++c) {
      const float xv = xr[c];
      o0 += xv * dw[c * 4 + 0];
      o1 += xv * dw[c * 4 + 1];
      o2 += xv * dw[c * 4 + 2];
      o3 += xv * dw[c * 4 + 3];
    }
    den[t * 4 + 0] = o0; den[t * 4 + 1] = o1;
    den[t * 4 + 2] = o2; den[t * 4 + 3] = o3;
  }
  __syncthreads();

  if (tid < 4) {                       // LI scan, one channel per thread
    float v = 0.0f;
    for (int t = 0; t < T_SZ; ++t) {
      v = v * 0.9f + den[t * 4 + tid] * 0.1f;
      den[t * 4 + tid] = v;
    }
  }
  __syncthreads();

  if (tid < 80) {                      // dp = mean(last12) - mean(first12)
    const int ch = tid & 3, g = tid >> 2;
    float s1 = 0.0f, s2 = 0.0f;
    for (int j = 0; j < 12; ++j) {
      s1 += den[(g * 24 + j) * 4 + ch];
      s2 += den[(g * 24 + 12 + j) * 4 + ch];
    }
    dp[g * 4 + ch] = (s2 - s1) * (1.0f / 12.0f);
  }
  __syncthreads();

  if (tid < 20) {                      // attention scores
    float sc = a2b[0];
    for (int j = 0; j < 8; ++j) {
      float h = a1b[j];
      for (int ch = 0; ch < 4; ++ch) h += dp[tid * 4 + ch] * a1w[ch * 8 + j];
      h = fmaxf(h, 0.0f);
      sc += h * a2w[j];
    }
    score[tid] = sc;
  }
  __syncthreads();

  if (tid == 0) {                      // softmax over 20 + weighted sum
    float mx = -1e30f;
    for (int g = 0; g < 20; ++g) mx = fmaxf(mx, score[g]);
    float se = 0.0f;
    for (int g = 0; g < 20; ++g) { score[g] = expf(score[g] - mx); se += score[g]; }
    const float inv = 1.0f / se;
    for (int ch = 0; ch < 4; ++ch) {
      float a = 0.0f;
      for (int g = 0; g < 20; ++g) a += dp[g * 4 + ch] * score[g];
      out[b * 4 + ch] = a * inv;
    }
  }
}

// --------------------------- host launcher ---------------------------------

extern "C" void kernel_launch(void* const* d_in, const int* in_sizes, int n_in,
                              void* d_out, int out_size, void* d_ws, size_t ws_size,
                              hipStream_t stream) {
  const float* x          = (const float*)d_in[0];
  const float* conv1_w    = (const float*)d_in[1];
  const float* bn1_scale  = (const float*)d_in[2];
  const float* bn1_bias   = (const float*)d_in[3];
  const float* bn1_mean   = (const float*)d_in[4];
  const float* bn1_var    = (const float*)d_in[5];
  const float* vth1_raw   = (const float*)d_in[6];
  const float* decay1_raw = (const float*)d_in[7];
  const float* conv2_w    = (const float*)d_in[8];
  const float* bn2_scale  = (const float*)d_in[9];
  const float* bn2_bias   = (const float*)d_in[10];
  const float* bn2_mean   = (const float*)d_in[11];
  const float* bn2_var    = (const float*)d_in[12];
  const float* vth2_raw   = (const float*)d_in[13];
  const float* decay2_raw = (const float*)d_in[14];
  const float* conv3_w    = (const float*)d_in[15];
  const float* bn3_scale  = (const float*)d_in[16];
  const float* bn3_bias   = (const float*)d_in[17];
  const float* bn3_mean   = (const float*)d_in[18];
  const float* bn3_var    = (const float*)d_in[19];
  const float* dense_w    = (const float*)d_in[20];
  const float* dense_b    = (const float*)d_in[21];
  const float* attn1_w    = (const float*)d_in[22];
  const float* attn1_b    = (const float*)d_in[23];
  const float* attn2_w    = (const float*)d_in[24];
  const float* attn2_b    = (const float*)d_in[25];

  // workspace layout (all sizes multiples of 256 bytes)
  char* ws = (char*)d_ws;
  size_t off = 0;
  __bf16* spk1p = (__bf16*)(ws + off); off += (size_t)B_SZ * TPAD * 64  * 2;
  __bf16* spk2p = (__bf16*)(ws + off); off += (size_t)B_SZ * TPAD * 128 * 2;
  float*  x1    = (float* )(ws + off); off += (size_t)B_SZ * T_SZ * 64  * 4;
  float*  x2    = (float* )(ws + off); off += (size_t)B_SZ * T_SZ * 128 * 4;
  float*  x3    = (float* )(ws + off); off += (size_t)B_SZ * T_SZ * 128 * 4;
  __bf16* w1t   = (__bf16*)(ws + off); off += 64 * 64 * 2;
  __bf16* w2t   = (__bf16*)(ws + off); off += 32 * 64 * 128 * 2;
  __bf16* w3t   = (__bf16*)(ws + off); off += 32 * 128 * 128 * 2;

  // output layout: logits | spk1 | spk2 | spk3 (flat, return order)
  float* out_logits = (float*)d_out;
  float* spk1_out = out_logits + (size_t)B_SZ * 4;
  float* spk2_out = spk1_out + (size_t)B_SZ * T_SZ * 64;
  float* spk3_out = spk2_out + (size_t)B_SZ * T_SZ * 128;

  // 1) zero padded spike buffers (spk1p/spk2p are contiguous)
  {
    const int n16 = (int)(((size_t)B_SZ * TPAD * 64 * 2 +
                           (size_t)B_SZ * TPAD * 128 * 2) / 16);
    zero_kernel<<<2048, 256, 0, stream>>>((uint4*)spk1p, n16);
  }
  // 2) weight conversion / transpose to bf16
  prep_weights<<<(32 * 128 * 128 + 255) / 256, 256, 0, stream>>>(
      conv1_w, conv2_w, conv3_w, w1t, w2t, w3t);

  // 3) conv1 + BN1 : 3840 waves, 8 waves per 256-thread block
  conv1_bn_wmma<<<480, 256, 0, stream>>>(x, w1t, bn1_scale, bn1_bias,
                                         bn1_mean, bn1_var, x1);
  // 4) hLIF stage 1
  hlif_kernel<<<(B_SZ * 64) / 256, 256, 0, stream>>>(
      x1, vth1_raw, decay1_raw, spk1_out, spk1p, 64);

  // 5) conv2 + BN2 : block per (b,t0)
  conv2_bn_wmma<<<B_SZ * 30, 256, 0, stream>>>(spk1p, w2t, bn2_scale, bn2_bias,
                                               bn2_mean, bn2_var, x2);
  // 6) hLIF stage 2
  hlif_kernel<<<(B_SZ * 128) / 256, 256, 0, stream>>>(
      x2, vth2_raw, decay2_raw, spk2_out, spk2p, 128);

  // 7) conv3 + BN3 : block per (b,t0)
  conv3_bn_wmma<<<B_SZ * 30, 256, 0, stream>>>(spk2p, w3t, bn3_scale, bn3_bias,
                                               bn3_mean, bn3_var, x3);
  // 8) ALIF stage 3
  alif_kernel<<<(B_SZ * 128) / 256, 256, 0, stream>>>(x3, spk3_out);

  // 9) fused readout
  readout_kernel<<<B_SZ, 128, 0, stream>>>(spk3_out, dense_w, dense_b,
                                           attn1_w, attn1_b, attn2_w, attn2_b,
                                           out_logits);
}
